// MyConditionalMultiHeadAttention_66451734004228
// MI455X (gfx1250) — compile-verified, matrix operand order checked by reading, not run on hardware
//
#include <hip/hip_runtime.h>

// MI455X / gfx1250 fused dual-stream MHA.
// Requires ws_size >= 60 MB (6 x 8MB f16 activation buffers + 6 x 2MB f16 W^T).
//
// d_out layout: [ out: B*LQ*D f32 | attn: H*B*LQ*LK f32 ]

#define D_   1024
#define LQ_  1024
#define LK_  1024
#define B_   4
#define H_   16
#define DK_  64
#define M_   (B_ * LQ_)   // 4096 rows for all projection GEMMs

typedef __attribute__((ext_vector_type(16))) _Float16 v16h;
typedef __attribute__((ext_vector_type(8)))  _Float16 v8h;
typedef __attribute__((ext_vector_type(8)))  float    v8f;
typedef __attribute__((ext_vector_type(4)))  float    v4f;

__device__ __forceinline__ v8f wmma16(v16h a, v16h b, v8f c) {
  // D = A(16x32 f16) * B(32x16 f16) + C(16x16 f32)
  return __builtin_amdgcn_wmma_f32_16x16x32_f16(false, a, false, b, (short)0, c,
                                                false, false);
}

// Fragment loader for f16 data stored k-contiguous per "row":
//  - A operand: row0 = M tile base, storage row-major [M][K]
//  - B operand: row0 = N tile base, storage "n-major"  [N][K]
// Per CDNA5 16-bit layout: lane holds row (lane&15); halves e0..e7 = k
// {g*8 .. g*8+7}, e8..e15 = k {16+g*8 ..}, g = lane>>4.  Two b128 loads.
__device__ __forceinline__ v16h frag_f16(const _Float16* __restrict__ base,
                                         int row0, int ld, int kbase, int lane) {
  const int g = (lane >> 4) & 1, r = lane & 15;
  const _Float16* p = base + (size_t)(row0 + r) * ld + kbase + g * 8;
  v8h c0 = *(const v8h*)(p);
  v8h c1 = *(const v8h*)(p + 16);
  v16h a;
#pragma unroll
  for (int i = 0; i < 8; ++i) { a[i] = c0[i]; a[8 + i] = c1[i]; }
  return a;
}

// Same fragment, but source is f32 row-major; convert on the fly.
__device__ __forceinline__ v16h frag_f32(const float* __restrict__ base,
                                         int row0, int ld, int kbase, int lane) {
  const int g = (lane >> 4) & 1, r = lane & 15;
  const float* p = base + (size_t)(row0 + r) * ld + kbase + g * 8;
  v4f c0 = *(const v4f*)(p);
  v4f c1 = *(const v4f*)(p + 4);
  v4f c2 = *(const v4f*)(p + 16);
  v4f c3 = *(const v4f*)(p + 20);
  v16h a;
#pragma unroll
  for (int i = 0; i < 4; ++i) {
    a[i]      = (_Float16)c0[i];
    a[4 + i]  = (_Float16)c1[i];
    a[8 + i]  = (_Float16)c2[i];
    a[12 + i] = (_Float16)c3[i];
  }
  return a;
}

// A-operand fragment from a 16x32 row-major f16 tile in LDS.
__device__ __forceinline__ v16h frag_lds(const _Float16* pt, int lane) {
  const int g = (lane >> 4) & 1, r = lane & 15;
  const _Float16* p = pt + r * 32 + g * 8;
  v8h c0 = *(const v8h*)(p);
  v8h c1 = *(const v8h*)(p + 16);
  v16h a;
#pragma unroll
  for (int i = 0; i < 8; ++i) { a[i] = c0[i]; a[8 + i] = c1[i]; }
  return a;
}

// ---------------------------------------------------------------- transpose W
// Wt[n][k] = (f16) W[k][n], 1024x1024.  grid (32,32), block (32,8).
__global__ __launch_bounds__(256) void wt_f16_kernel(const float* __restrict__ W,
                                                     _Float16* __restrict__ Wt) {
  __shared__ float t[32][33];
  const int tx = threadIdx.x, ty = threadIdx.y;
  const int bx = blockIdx.x * 32, by = blockIdx.y * 32;
#pragma unroll
  for (int j = 0; j < 32; j += 8)
    t[ty + j][tx] = W[(size_t)(by + ty + j) * D_ + bx + tx];
  __syncthreads();
#pragma unroll
  for (int j = 0; j < 32; j += 8)
    Wt[(size_t)(bx + ty + j) * D_ + by + tx] = (_Float16)t[tx][ty + j];
}

// --------------------------------------------------------------- projections
// Y = X(f32 [4096,1024]) @ Wt^T + bias -> f16.  One wave per 16x64 C tile.
// vtrans==0: Y row-major [M][D].  vtrans==1 (V proj): Y = [B][H][DK][LK].
__global__ __launch_bounds__(32) void proj_kernel(const float* __restrict__ X,
                                                  const _Float16* __restrict__ Wt,
                                                  const float* __restrict__ bias,
                                                  _Float16* __restrict__ Y,
                                                  int vtrans) {
  const int lane = threadIdx.x;
  const int n0 = blockIdx.x * 64, m0 = blockIdx.y * 16;
  v8f acc[4] = {};
  for (int k = 0; k < D_; k += 32) {
    v16h a = frag_f32(X, m0, D_, k, lane);
#pragma unroll
    for (int t = 0; t < 4; ++t) {
      v16h bfr = frag_f16(Wt, n0 + t * 16, D_, k, lane);
      acc[t] = wmma16(a, bfr, acc[t]);
    }
  }
  const int g = lane >> 4, c = lane & 15;
#pragma unroll
  for (int t = 0; t < 4; ++t) {
    const int n = n0 + t * 16 + c;
    const float bv = bias[n];
#pragma unroll
    for (int r = 0; r < 8; ++r) {
      const float vv = acc[t][r] + bv;
      const int m = m0 + g * 8 + r;
      if (!vtrans) {
        Y[(size_t)m * D_ + n] = (_Float16)vv;
      } else {
        const int bb = m >> 10, lk = m & 1023;   // m = b*LK + lk
        const int hh = n >> 6,  dk = n & 63;     // n = h*DK + dk
        Y[(((size_t)bb * H_ + hh) * DK_ + dk) * LK_ + lk] = (_Float16)vv;
      }
    }
  }
}

// ---------------------------------------------------------------- final GEMM
// out = OH(f16 [4096,1024]) @ Wo + bo -> f32 d_out.
__global__ __launch_bounds__(32) void out_gemm_kernel(const _Float16* __restrict__ X,
                                                      const _Float16* __restrict__ Wt,
                                                      const float* __restrict__ bias,
                                                      float* __restrict__ Y) {
  const int lane = threadIdx.x;
  const int n0 = blockIdx.x * 64, m0 = blockIdx.y * 16;
  v8f acc[4] = {};
  for (int k = 0; k < D_; k += 32) {
    v16h a = frag_f16(X, m0, D_, k, lane);
#pragma unroll
    for (int t = 0; t < 4; ++t) {
      v16h bfr = frag_f16(Wt, n0 + t * 16, D_, k, lane);
      acc[t] = wmma16(a, bfr, acc[t]);
    }
  }
  const int g = lane >> 4, c = lane & 15;
#pragma unroll
  for (int t = 0; t < 4; ++t) {
    const int n = n0 + t * 16 + c;
    const float bv = bias[n];
#pragma unroll
    for (int r = 0; r < 8; ++r)
      Y[(size_t)(m0 + g * 8 + r) * D_ + n] = acc[t][r] + bv;
  }
}

// ---------------------------------------------------------- fused attention
// One wave handles a 16-row query strip of one (b,h).
// Pass 1: online softmax stats over all 64 key tiles (dual-stream QK, K=128).
// Pass 2: recompute scores, write NORMALIZED attn to [H][B][LQ][LK], and
//         accumulate P@V via an LDS-restaged 16x32 P tile (C-layout -> A-layout).
__global__ __launch_bounds__(32) void attn_kernel(const _Float16* __restrict__ q1h,
                                                  const _Float16* __restrict__ q2h,
                                                  const _Float16* __restrict__ k1h,
                                                  const _Float16* __restrict__ k2h,
                                                  const _Float16* __restrict__ vt,
                                                  const unsigned char* __restrict__ mask,
                                                  float* __restrict__ attn,
                                                  _Float16* __restrict__ oh) {
  __shared__ __align__(16) _Float16 pt[16 * 32];
  const int lane = threadIdx.x;
  const int g = lane >> 4, cidx = lane & 15;
  const int m0 = blockIdx.x * 16;
  const int h = blockIdx.y, b = blockIdx.z;
  const float scale = 0.125f;  // 1/sqrt(64)

  const _Float16* q1b = q1h + (size_t)b * LQ_ * D_ + h * DK_;
  const _Float16* q2b = q2h + (size_t)b * LQ_ * D_ + h * DK_;
  const _Float16* k1b = k1h + (size_t)b * LK_ * D_ + h * DK_;
  const _Float16* k2b = k2h + (size_t)b * LK_ * D_ + h * DK_;
  const _Float16* vtb = vt + ((size_t)b * H_ + h) * DK_ * LK_;
  const unsigned char* mb = mask + (size_t)b * LK_;
  float* ap = attn + ((size_t)h * B_ + b) * (size_t)LQ_ * LK_;

  // Q fragments (128-dim split as q1[0:32],q1[32:64],q2[0:32],q2[32:64])
  const v16h aq0 = frag_f16(q1b, m0, D_, 0, lane);
  const v16h aq1 = frag_f16(q1b, m0, D_, 32, lane);
  const v16h aq2 = frag_f16(q2b, m0, D_, 0, lane);
  const v16h aq3 = frag_f16(q2b, m0, D_, 32, lane);

  float mrow[8], lrow[8];
#pragma unroll
  for (int r = 0; r < 8; ++r) { mrow[r] = -__builtin_inff(); lrow[r] = 0.f; }

  // ----- pass 1: per-lane online max / sum-of-exp over this lane's columns
  for (int t = 0; t < LK_ / 16; ++t) {
    const int n0 = t * 16;
    v8f c = {};
    c = wmma16(aq0, frag_f16(k1b, n0, D_, 0, lane), c);
    c = wmma16(aq1, frag_f16(k1b, n0, D_, 32, lane), c);
    c = wmma16(aq2, frag_f16(k2b, n0, D_, 0, lane), c);
    c = wmma16(aq3, frag_f16(k2b, n0, D_, 32, lane), c);
    const bool mk = mb[n0 + cidx] != 0;
#pragma unroll
    for (int r = 0; r < 8; ++r) {
      const float s = mk ? -__builtin_inff() : c[r] * scale;
      const float mn = fmaxf(mrow[r], s);
      const float e = __expf(fminf(mrow[r], s) - mn);  // one exp per update
      lrow[r] = (s > mrow[r]) ? (lrow[r] * e + 1.0f) : (lrow[r] + e);
      mrow[r] = mn;
    }
  }

  // ----- combine (m,l) across the 16-lane group holding each row
#pragma unroll
  for (int r = 0; r < 8; ++r) {
#pragma unroll
    for (int off = 1; off < 16; off <<= 1) {
      const float mo = __shfl_xor(mrow[r], off, 32);
      const float lo = __shfl_xor(lrow[r], off, 32);
      const float mn = fmaxf(mrow[r], mo);
      lrow[r] = lrow[r] * __expf(mrow[r] - mn) + lo * __expf(mo - mn);
      mrow[r] = mn;
    }
    lrow[r] = 1.0f / lrow[r];  // reuse as reciprocal of row sum
  }

  // ----- pass 2: write normalized attn + accumulate P @ V
  v8f co[4] = {};
  for (int tp = 0; tp < LK_ / 32; ++tp) {
#pragma unroll
    for (int sub = 0; sub < 2; ++sub) {
      const int n0 = (tp * 2 + sub) * 16;
      v8f c = {};
      c = wmma16(aq0, frag_f16(k1b, n0, D_, 0, lane), c);
      c = wmma16(aq1, frag_f16(k1b, n0, D_, 32, lane), c);
      c = wmma16(aq2, frag_f16(k2b, n0, D_, 0, lane), c);
      c = wmma16(aq3, frag_f16(k2b, n0, D_, 32, lane), c);
      const bool mk = mb[n0 + cidx] != 0;
#pragma unroll
      for (int r = 0; r < 8; ++r) {
        const float s = mk ? -__builtin_inff() : c[r] * scale;
        const float p = __expf(s - mrow[r]) * lrow[r];
        const int row = g * 8 + r;
        ap[(size_t)(m0 + row) * LK_ + n0 + cidx] = p;
        pt[row * 32 + sub * 16 + cidx] = (_Float16)p;  // stage for A-layout
      }
    }
    // same-wave LDS ops are in-order: P tile is visible to the loads below
    const v16h pa = frag_lds(pt, lane);
    const int kb = tp * 32;
#pragma unroll
    for (int dt = 0; dt < 4; ++dt) {
      const v16h bv = frag_f16(vtb, dt * 16, LK_, kb, lane);
      co[dt] = wmma16(pa, bv, co[dt]);
    }
  }

  // ----- store out_head [B][LQ][H*DK] (f16, feeds final GEMM)
#pragma unroll
  for (int dt = 0; dt < 4; ++dt) {
#pragma unroll
    for (int r = 0; r < 8; ++r) {
      const int m = m0 + g * 8 + r;
      const int dk = dt * 16 + cidx;
      oh[((size_t)b * LQ_ + m) * D_ + h * DK_ + dk] = (_Float16)co[dt][r];
    }
  }
}

// ------------------------------------------------------------------- launch
extern "C" void kernel_launch(void* const* d_in, const int* in_sizes, int n_in,
                              void* d_out, int out_size, void* d_ws, size_t ws_size,
                              hipStream_t stream) {
  const float* q1 = (const float*)d_in[0];
  const float* q2 = (const float*)d_in[1];
  const float* k1 = (const float*)d_in[2];
  const float* k2 = (const float*)d_in[3];
  const float* v  = (const float*)d_in[4];
  const unsigned char* mask = (const unsigned char*)d_in[5];  // jax bool -> 1B
  const float* Wq  = (const float*)d_in[6];
  const float* bq  = (const float*)d_in[7];
  const float* Wk  = (const float*)d_in[8];
  const float* bk  = (const float*)d_in[9];
  const float* Wq2 = (const float*)d_in[10];
  const float* bq2 = (const float*)d_in[11];
  const float* Wk2 = (const float*)d_in[12];
  const float* bk2 = (const float*)d_in[13];
  const float* Wv  = (const float*)d_in[14];
  const float* bv  = (const float*)d_in[15];
  const float* Wo  = (const float*)d_in[16];
  const float* bo  = (const float*)d_in[17];

  float* out  = (float*)d_out;
  float* attn = out + (size_t)B_ * LQ_ * D_;

  char* w = (char*)d_ws;
  const size_t SZA = (size_t)M_ * D_ * sizeof(_Float16);   // 8 MB
  const size_t SZW = (size_t)D_ * D_ * sizeof(_Float16);   // 2 MB
  _Float16* q1h = (_Float16*)(w + 0 * SZA);
  _Float16* q2h = (_Float16*)(w + 1 * SZA);
  _Float16* k1h = (_Float16*)(w + 2 * SZA);
  _Float16* k2h = (_Float16*)(w + 3 * SZA);
  _Float16* vt  = (_Float16*)(w + 4 * SZA);   // [B][H][DK][LK]
  _Float16* oh  = (_Float16*)(w + 5 * SZA);   // [B][LQ][D]
  char* wt = w + 6 * SZA;
  _Float16* WtQ  = (_Float16*)(wt + 0 * SZW);
  _Float16* WtK  = (_Float16*)(wt + 1 * SZW);
  _Float16* WtQ2 = (_Float16*)(wt + 2 * SZW);
  _Float16* WtK2 = (_Float16*)(wt + 3 * SZW);
  _Float16* WtV  = (_Float16*)(wt + 4 * SZW);
  _Float16* WtO  = (_Float16*)(wt + 5 * SZW);

  // 1) transpose + f16-convert all weights
  {
    dim3 tg(32, 32), tb(32, 8);
    wt_f16_kernel<<<tg, tb, 0, stream>>>(Wq,  WtQ);
    wt_f16_kernel<<<tg, tb, 0, stream>>>(Wk,  WtK);
    wt_f16_kernel<<<tg, tb, 0, stream>>>(Wq2, WtQ2);
    wt_f16_kernel<<<tg, tb, 0, stream>>>(Wk2, WtK2);
    wt_f16_kernel<<<tg, tb, 0, stream>>>(Wv,  WtV);
    wt_f16_kernel<<<tg, tb, 0, stream>>>(Wo,  WtO);
  }

  // 2) projections
  {
    dim3 pg(D_ / 64, M_ / 16);  // (16, 256) -> one wave per 16x64 tile
    proj_kernel<<<pg, 32, 0, stream>>>(q1, WtQ,  bq,  q1h, 0);
    proj_kernel<<<pg, 32, 0, stream>>>(q2, WtQ2, bq2, q2h, 0);
    proj_kernel<<<pg, 32, 0, stream>>>(k1, WtK,  bk,  k1h, 0);
    proj_kernel<<<pg, 32, 0, stream>>>(k2, WtK2, bk2, k2h, 0);
    proj_kernel<<<pg, 32, 0, stream>>>(v,  WtV,  bv,  vt,  1);
  }

  // 3) fused attention (writes normalized attn + out_head)
  {
    dim3 ag(LQ_ / 16, H_, B_);
    attn_kernel<<<ag, 32, 0, stream>>>(q1h, q2h, k1h, k2h, vt, mask, attn, oh);
  }

  // 4) output projection
  {
    dim3 pg(D_ / 64, M_ / 16);
    out_gemm_kernel<<<pg, 32, 0, stream>>>(oh, WtO, bo, out);
  }
}
